// VectorQuantizer_26551487824076
// MI455X (gfx1250) — compile-verified, matrix operand order checked by reading, not run on hardware
//
#include <hip/hip_runtime.h>

// ---------------------------------------------------------------------------
// VQ-VAE vector quantizer for MI455X (gfx1250, wave32).
// dist(x,e) = ||x||^2 + ||e||^2 - 2 x.e ; argmin unaffected by ||x||^2.
// GEMM via split-bf16 (bf16x3) V_WMMA_F32_16X16X32_BF16; codebook streamed
// through LDS with GLOBAL_LOAD_ASYNC_TO_LDS_B128 double buffering (ASYNCcnt).
// ---------------------------------------------------------------------------

typedef __bf16 v16bf __attribute__((ext_vector_type(16)));
typedef float  v8f   __attribute__((ext_vector_type(8)));

#define DVQ     256            // embedding dim
#define KCODES  8192           // codebook size
#define NROWS   32768          // 32*1024 flattened inputs
#define CHUNK   64             // codes staged in LDS per iteration
#define NCHUNKS (KCODES / CHUNK)
#define NTILES  (CHUNK / 16)   // 16-wide N tiles per chunk
#define KT      (DVQ / 32)     // 32-deep K tiles (bf16 WMMA K=32) -> 8

// CDNA5 async memory->LDS copy, tracked by ASYNCcnt (no VGPR data movement).
__device__ __forceinline__ void async_copy16(const uint4* g, uint4* l) {
    const unsigned lds_off = (unsigned)(uintptr_t)l;  // addr[31:0] = LDS offset
    asm volatile("global_load_async_to_lds_b128 %0, %1, off"
                 :: "v"(lds_off), "v"(g)
                 : "memory");
}
__device__ __forceinline__ void wait_async0() {
    asm volatile("s_wait_asynccnt 0" ::: "memory");
}

// ---------------------------------------------------------------------------
// Kernel 1: convert codebook to split-bf16 (hi + lo), compute ||e||^2,
// zero the loss accumulator. One wave per codebook row.
// ---------------------------------------------------------------------------
__global__ void vq_prep(const float* __restrict__ E,
                        __bf16* __restrict__ Ehi,
                        __bf16* __restrict__ Elo,
                        float* __restrict__ norms,
                        float* __restrict__ lossPtr) {
    if (blockIdx.x == 0 && threadIdx.x == 0) *lossPtr = 0.0f;

    const int wavesPerBlock = blockDim.x >> 5;
    const int wave = blockIdx.x * wavesPerBlock + (threadIdx.x >> 5);
    const int lane = threadIdx.x & 31;
    if (wave >= KCODES) return;

    const float* row = E + (size_t)wave * DVQ;
    float s = 0.0f;
#pragma unroll
    for (int j = 0; j < 8; ++j) {               // 256 dims / 32 lanes
        const int i = lane * 8 + j;
        const float x = row[i];
        const __bf16 h = (__bf16)x;
        Ehi[(size_t)wave * DVQ + i] = h;
        Elo[(size_t)wave * DVQ + i] = (__bf16)(x - (float)h);
        s += x * x;
    }
#pragma unroll
    for (int off = 16; off > 0; off >>= 1) s += __shfl_xor(s, off, 32);
    if (lane == 0) norms[wave] = s;
}

// ---------------------------------------------------------------------------
// Kernel 2: main quantizer. 512 threads = 16 waves, each wave owns 16 rows.
// Block covers 256 rows; grid = 128 blocks. Codebook double-buffered in LDS.
// ---------------------------------------------------------------------------
__global__ __launch_bounds__(512, 1)
void vq_main(const float* __restrict__ X,
             const float* __restrict__ E,
             const __bf16* __restrict__ Ehi,
             const __bf16* __restrict__ Elo,
             const float* __restrict__ norms,
             float* __restrict__ outQ,
             float* __restrict__ outLoss,
             float* __restrict__ outIdx) {
    __shared__ __align__(16) __bf16 sEhi[2][CHUNK * DVQ];   // 2 x 32 KB
    __shared__ __align__(16) __bf16 sElo[2][CHUNK * DVQ];   // 2 x 32 KB
    __shared__ int   sIdx[256];
    __shared__ float sErr[16];

    const int tid  = threadIdx.x;
    const int wave = tid >> 5;
    const int lane = tid & 31;
    const int half = lane >> 4;     // which K-subset this lane holds (A/B layout)
    const int l16  = lane & 15;

    const int gRowBase = blockIdx.x * 256;
    const int row      = gRowBase + wave * 16 + l16;

    // ---- Kick off async staging of chunk 0 into buffer 0 (overlaps with
    //      the A-fragment build below). 64 KB / 512 threads = 8 x b128 each.
    {
        const uint4* gh = (const uint4*)Ehi;
        const uint4* gl = (const uint4*)Elo;
        uint4* sh = (uint4*)sEhi[0];
        uint4* sl = (uint4*)sElo[0];
#pragma unroll
        for (int i = 0; i < 4; ++i) {
            async_copy16(gh + tid + i * 512, sh + tid + i * 512);
            async_copy16(gl + tid + i * 512, sl + tid + i * 512);
        }
    }

    // ---- Build split-bf16 A fragments for this wave's 16 rows (once). ----
    // ISA A-layout (16-bit, 16x32): lane L (<16) row=L, K={0..7,16..23};
    // lane L+16 row=L, K={8..15,24..31} -> two contiguous 8-float spans.
    v16bf ahi[KT], alo[KT];
    const float* xrow = X + (size_t)row * DVQ;
#pragma unroll
    for (int kt = 0; kt < KT; ++kt) {
        const int b0 = kt * 32 + half * 8;
        float t[16];
        const float4 f0 = *(const float4*)(xrow + b0);
        const float4 f1 = *(const float4*)(xrow + b0 + 4);
        const float4 f2 = *(const float4*)(xrow + b0 + 16);
        const float4 f3 = *(const float4*)(xrow + b0 + 20);
        t[0]=f0.x; t[1]=f0.y; t[2]=f0.z; t[3]=f0.w;
        t[4]=f1.x; t[5]=f1.y; t[6]=f1.z; t[7]=f1.w;
        t[8]=f2.x; t[9]=f2.y; t[10]=f2.z; t[11]=f2.w;
        t[12]=f3.x; t[13]=f3.y; t[14]=f3.z; t[15]=f3.w;
#pragma unroll
        for (int i = 0; i < 16; ++i) {
            const __bf16 h = (__bf16)t[i];
            ahi[kt][i] = h;
            alo[kt][i] = (__bf16)(t[i] - (float)h);
        }
    }

    // Running (min, argmin) for the 8 accumulator rows this lane holds.
    float minv[8];
    int   mini[8];
#pragma unroll
    for (int v = 0; v < 8; ++v) { minv[v] = 3.4e38f; mini[v] = 0; }

    wait_async0();          // chunk 0 resident (this wave's copies)
    __syncthreads();        // ... and everyone else's

    // ---- Stream the codebook: compute chunk c while async-staging c+1. ----
    for (int chunk = 0; chunk < NCHUNKS; ++chunk) {
        const int cur = chunk & 1;

        if (chunk + 1 < NCHUNKS) {      // stage next chunk into other buffer
            const uint4* gh = (const uint4*)(Ehi + (size_t)(chunk + 1) * CHUNK * DVQ);
            const uint4* gl = (const uint4*)(Elo + (size_t)(chunk + 1) * CHUNK * DVQ);
            uint4* sh = (uint4*)sEhi[1 - cur];
            uint4* sl = (uint4*)sElo[1 - cur];
#pragma unroll
            for (int i = 0; i < 4; ++i) {
                async_copy16(gh + tid + i * 512, sh + tid + i * 512);
                async_copy16(gl + tid + i * 512, sl + tid + i * 512);
            }
        }

#pragma unroll
        for (int nt = 0; nt < NTILES; ++nt) {
            v8f c = {};
            const int codeLocal = nt * 16 + l16;   // B-layout: lane n holds col n
            const __bf16* bh_base = sEhi[cur] + codeLocal * DVQ;
            const __bf16* bl_base = sElo[cur] + codeLocal * DVQ;
#pragma unroll
            for (int kt = 0; kt < KT; ++kt) {
                const int b0 = kt * 32 + half * 8;
                union { v16bf v; uint4 q[2]; } bh, bl;
                bh.q[0] = *(const uint4*)(bh_base + b0);
                bh.q[1] = *(const uint4*)(bh_base + b0 + 16);
                bl.q[0] = *(const uint4*)(bl_base + b0);
                bl.q[1] = *(const uint4*)(bl_base + b0 + 16);
                // bf16x3: hi*hi + lo*hi + hi*lo (near-fp32 dot product)
                c = __builtin_amdgcn_wmma_f32_16x16x32_bf16(
                        false, ahi[kt], false, bh.v, (short)0, c, false, false);
                c = __builtin_amdgcn_wmma_f32_16x16x32_bf16(
                        false, alo[kt], false, bh.v, (short)0, c, false, false);
                c = __builtin_amdgcn_wmma_f32_16x16x32_bf16(
                        false, ahi[kt], false, bl.v, (short)0, c, false, false);
            }
            const int   code = chunk * CHUNK + nt * 16 + l16;
            const float en   = norms[code];        // 32 KB total: L2-hot
#pragma unroll
            for (int v = 0; v < 8; ++v) {
                const float d = en - 2.0f * c[v];
                if (d < minv[v]) { minv[v] = d; mini[v] = code; }
            }
        }

        wait_async0();      // next buffer fully written by this wave
        __syncthreads();    // all waves done writing next / reading cur
    }

    // ---- Argmin reduce across the 16 lanes sharing each accumulator row. ----
    // C/D layout: lane<16 -> row v, lane>=16 -> row v+8 (N = lane%16).
#pragma unroll
    for (int v = 0; v < 8; ++v) {
        float mv = minv[v];
        int   mi = mini[v];
#pragma unroll
        for (int off = 1; off < 16; off <<= 1) {
            const float ov = __shfl_xor(mv, off, 32);
            const int   oi = __shfl_xor(mi, off, 32);
            if (ov < mv || (ov == mv && oi < mi)) { mv = ov; mi = oi; }
        }
        if (l16 == 0) sIdx[wave * 16 + half * 8 + v] = mi;
    }
    __syncthreads();

    // ---- Epilogue: indices, gather quantized rows, loss partials. ----
    if (tid < 256) outIdx[gRowBase + tid] = (float)sIdx[tid];

    const int r    = tid >> 1;          // 2 threads per output row
    const int part = tid & 1;
    const int code = sIdx[r];
    const float* eR = E + (size_t)code * DVQ + part * 128;
    const float* xR = X + (size_t)(gRowBase + r) * DVQ + part * 128;
    float* qR = outQ + (size_t)(gRowBase + r) * DVQ + part * 128;
    float err = 0.0f;
#pragma unroll 4
    for (int i = 0; i < 128; i += 4) {
        const float4 e4 = *(const float4*)(eR + i);
        const float4 x4 = *(const float4*)(xR + i);
        *(float4*)(qR + i) = e4;   // straight-through forward value == quantized
        const float dx = e4.x - x4.x, dy = e4.y - x4.y;
        const float dz = e4.z - x4.z, dw = e4.w - x4.w;
        err += dx * dx + dy * dy + dz * dz + dw * dw;
    }
#pragma unroll
    for (int off = 16; off > 0; off >>= 1) err += __shfl_xor(err, off, 32);
    if (lane == 0) sErr[wave] = err;
    __syncthreads();
    if (tid == 0) {
        float tot = 0.0f;
#pragma unroll
        for (int w = 0; w < 16; ++w) tot += sErr[w];
        // loss = q_latent + COMMITMENT_COST * e_latent = 2 * mean((q-x)^2)
        atomicAdd(outLoss, tot * (2.0f / ((float)NROWS * (float)DVQ)));
    }
}

// ---------------------------------------------------------------------------
extern "C" void kernel_launch(void* const* d_in, const int* in_sizes, int n_in,
                              void* d_out, int out_size, void* d_ws, size_t ws_size,
                              hipStream_t stream) {
    (void)in_sizes; (void)n_in; (void)out_size; (void)ws_size;
    const float* X = (const float*)d_in[0];   // (32,1024,256) f32
    const float* E = (const float*)d_in[1];   // (8192,256) f32

    // workspace layout: Ehi (4MB) | Elo (4MB) | norms (32KB)
    __bf16* Ehi  = (__bf16*)d_ws;
    __bf16* Elo  = Ehi + (size_t)KCODES * DVQ;
    float*  norms = (float*)(Elo + (size_t)KCODES * DVQ);

    // d_out layout: quantized_st (32768*256) | loss (1) | indices (32768)
    float* outQ    = (float*)d_out;
    float* outLoss = outQ + (size_t)NROWS * DVQ;
    float* outIdx  = outLoss + 1;

    vq_prep<<<KCODES / 8, 256, 0, stream>>>(E, Ehi, Elo, norms, outLoss);
    vq_main<<<NROWS / 256, 512, 0, stream>>>(X, E, Ehi, Elo, norms,
                                             outQ, outLoss, outIdx);
}
